// SelfAttention2D_8237747273943
// MI455X (gfx1250) — matrix operand order repaired and checked by reading, not compiled, not run on hardware
//
#include <hip/hip_runtime.h>
#include <hip/hip_bf16.h>
#include <math.h>

typedef __attribute__((ext_vector_type(16))) _Float16 v16h;
typedef __attribute__((ext_vector_type(8)))  _Float16 v8h;
typedef __attribute__((ext_vector_type(8)))  float    v8f;

#define B_  4
#define C_  256
#define N_  4096
#define CQ_ 32

// ---------------------------------------------------------------------------
// Kernel 1: fused 1x1-conv projections. Emits q,k as (B,N,Cq) f16 row-major,
// v as (B,C,N) f16 (natural layout). All WMMA fragments in kernel 2 become
// contiguous vector loads because of these layouts.
// ---------------------------------------------------------------------------
__global__ __launch_bounds__(256) void proj_kernel(
    const float* __restrict__ x,
    const float* __restrict__ Wq, const float* __restrict__ bq,
    const float* __restrict__ Wk, const float* __restrict__ bk,
    const float* __restrict__ Wv, const float* __restrict__ bv,
    _Float16* __restrict__ qh, _Float16* __restrict__ kh, _Float16* __restrict__ vh)
{
    __shared__ float xs[C_ * 64];                 // 64 KB: x[b, :, n0..n0+63]
    const int b  = blockIdx.y;
    const int n0 = blockIdx.x * 64;
    const int t  = threadIdx.x;

    // stage x tile (float4, fully coalesced)
    for (int i = 0; i < 16; ++i) {
        int f4  = t + i * 256;                    // 0..4095 float4s
        int idx = f4 * 4;
        int c   = idx >> 6;
        int pix = idx & 63;
        *(float4*)&xs[c * 64 + pix] =
            *(const float4*)(x + ((size_t)(b * C_ + c)) * N_ + n0 + pix);
    }
    __syncthreads();

    const int pix = t & 63;
    const int og  = t >> 6;                       // 0..3 (uniform per wave)
    const int n   = n0 + pix;

    // q,k projections: 8 output channels per thread
    for (int oo = 0; oo < 8; ++oo) {
        int o = og * 8 + oo;
        float aq = bq[o], ak = bk[o];
        const float* wq = Wq + (size_t)o * C_;
        const float* wk = Wk + (size_t)o * C_;
        #pragma unroll 8
        for (int c = 0; c < C_; ++c) {
            float xv = xs[c * 64 + pix];          // bank = pix -> conflict-free
            aq = fmaf(wq[c], xv, aq);
            ak = fmaf(wk[c], xv, ak);
        }
        qh[((size_t)(b * N_ + n)) * CQ_ + o] = (_Float16)aq;
        kh[((size_t)(b * N_ + n)) * CQ_ + o] = (_Float16)ak;
    }

    // v projection: 64 output channels per thread, 4 at a time
    for (int ob = 0; ob < 64; ob += 4) {
        int o = og * 64 + ob;
        float a0 = bv[o], a1 = bv[o + 1], a2 = bv[o + 2], a3 = bv[o + 3];
        const float* w = Wv + (size_t)o * C_;
        #pragma unroll 4
        for (int c = 0; c < C_; ++c) {
            float xv = xs[c * 64 + pix];
            a0 = fmaf(w[c],          xv, a0);
            a1 = fmaf(w[C_ + c],     xv, a1);
            a2 = fmaf(w[2 * C_ + c], xv, a2);
            a3 = fmaf(w[3 * C_ + c], xv, a3);
        }
        vh[((size_t)(b * C_ + o    )) * N_ + n] = (_Float16)a0;
        vh[((size_t)(b * C_ + o + 1)) * N_ + n] = (_Float16)a1;
        vh[((size_t)(b * C_ + o + 2)) * N_ + n] = (_Float16)a2;
        vh[((size_t)(b * C_ + o + 3)) * N_ + n] = (_Float16)a3;
    }
}

// ---------------------------------------------------------------------------
// Kernel 2: flash-attention, one wave32 per (batch, 16-query tile).
// S^T = K_tile x Q^T so that exp(S^T) lands directly in A-matrix layout for
// the P x V^T WMMAs (no transpose). Pass A: global row max only. Pass B:
// software-pipelined by one 32-key chunk: QK WMMAs for chunk i+1 issue first,
// then the 16 independent PV WMMAs for chunk i keep the matrix pipe busy
// while QK executes; exps/pack for i+1 run last (result ready, co-executes).
// ---------------------------------------------------------------------------
__global__ __launch_bounds__(32) void attn_kernel(
    const _Float16* __restrict__ qh, const _Float16* __restrict__ kh,
    const _Float16* __restrict__ vh, const float* __restrict__ x,
    const float* __restrict__ gamma_p, float* __restrict__ out)
{
    __shared__ float stage[16 * 16];
    const int lane = threadIdx.x;
    const int half = lane >> 4;
    const int l16  = lane & 15;
    const int b    = blockIdx.y;
    const int q0   = blockIdx.x * 16;

    // Q^T B-fragment (loop invariant): lane holds q[q0+l16][16*half .. +15]
    const v16h qB = *(const v16h*)(qh + ((size_t)(b * N_ + q0 + l16)) * CQ_ + half * 16);

    const _Float16* kbase = kh + (size_t)b * N_ * CQ_;
    const v8f zero = {};

    // S^T for one 32-key chunk: two 16x16 logit tiles (keys x queries)
    auto qk = [&](int m0, v8f& s0, v8f& s1) {
        const _Float16* k0 = kbase + (size_t)(m0 + l16) * CQ_ + half * 16;
        __builtin_prefetch(k0 + 32 * CQ_, 0, 3);
        v8h a0lo = *(const v8h*)(k0);
        v8h a0hi = *(const v8h*)(k0 + 8);
        const _Float16* k1 = k0 + 16 * CQ_;
        v8h a1lo = *(const v8h*)(k1);
        v8h a1hi = *(const v8h*)(k1 + 8);
        v16h A0 = __builtin_shufflevector(a0lo, a0hi, 0,1,2,3,4,5,6,7,8,9,10,11,12,13,14,15);
        v16h A1 = __builtin_shufflevector(a1lo, a1hi, 0,1,2,3,4,5,6,7,8,9,10,11,12,13,14,15);
        s0 = __builtin_amdgcn_wmma_f32_16x16x32_f16(false, A0, false, qB, (short)0, zero, false, false);
        s1 = __builtin_amdgcn_wmma_f32_16x16x32_f16(false, A1, false, qB, (short)0, zero, false, false);
    };

    // ---- pass A: global max over this lane's key subset (pairwise trees)
    float mr = -1e30f;
    for (int m0 = 0; m0 < N_; m0 += 32) {
        v8f s0, s1;
        qk(m0, s0, s1);
        float a = fmaxf(fmaxf(s0[0], s0[1]), fmaxf(s0[2], s0[3]));
        float c = fmaxf(fmaxf(s0[4], s0[5]), fmaxf(s0[6], s0[7]));
        float d = fmaxf(fmaxf(s1[0], s1[1]), fmaxf(s1[2], s1[3]));
        float e = fmaxf(fmaxf(s1[4], s1[5]), fmaxf(s1[6], s1[7]));
        mr = fmaxf(mr, fmaxf(fmaxf(a, c), fmaxf(d, e)));
    }
    // merge lane halves: lane L and L^16 hold the same query column
    mr = fmaxf(mr, __shfl_xor(mr, 16, 32));

    // e = exp(s - m): accumulate l-sum and produce the P A-fragment
    float lr = 0.f;
    auto softmaxize = [&](const v8f& s0, const v8f& s1) -> v16h {
        float e0[8], e1[8];
        v16h pA;
        #pragma unroll
        for (int r = 0; r < 8; ++r) { e0[r] = __expf(s0[r] - mr); pA[r]     = (_Float16)e0[r]; }
        #pragma unroll
        for (int r = 0; r < 8; ++r) { e1[r] = __expf(s1[r] - mr); pA[8 + r] = (_Float16)e1[r]; }
        float t0 = (e0[0] + e0[1]) + (e0[2] + e0[3]);
        float t1 = (e0[4] + e0[5]) + (e0[6] + e0[7]);
        float t2 = (e1[0] + e1[1]) + (e1[2] + e1[3]);
        float t3 = (e1[4] + e1[5]) + (e1[6] + e1[7]);
        lr += (t0 + t1) + (t2 + t3);
        return pA;
    };

    // ---- pass B: software-pipelined PV accumulation over 256 channels
    v8f acc[16];
    #pragma unroll
    for (int c = 0; c < 16; ++c) acc[c] = zero;
    const _Float16* vbase = vh + (size_t)b * C_ * N_;

    v8f s0, s1;
    qk(0, s0, s1);
    v16h pA = softmaxize(s0, s1);

    for (int m0 = 0; m0 < N_ - 32; m0 += 32) {
        // (1) QK for chunk i+1: loads + 2 WMMAs enter the pipe first
        qk(m0 + 32, s0, s1);
        __builtin_prefetch(vbase + (size_t)l16 * N_ + m0 + 32, 0, 3);
        // (2) PV for chunk i: 16 independent WMMAs, no dependence on s0/s1
        #pragma unroll
        for (int ct = 0; ct < 16; ++ct) {
            const _Float16* vp = vbase + (size_t)(ct * 16 + l16) * N_ + m0 + half * 8;
            const v16h vB = __builtin_shufflevector(*(const v8h*)(vp), *(const v8h*)(vp + 16), 0,1,2,3,4,5,6,7,8,9,10,11,12,13,14,15);
            acc[ct] = __builtin_amdgcn_wmma_f32_16x16x32_f16(false, pA, false, vB, (short)0, acc[ct], false, false);
        }
        // (3) exps/pack for chunk i+1 (QK result ready by now)
        pA = softmaxize(s0, s1);
    }
    // epilogue: PV for the final chunk
    #pragma unroll
    for (int ct = 0; ct < 16; ++ct) {
        const _Float16* vp = vbase + (size_t)(ct * 16 + l16) * N_ + (N_ - 32) + half * 8;
        const v16h vB = __builtin_shufflevector(*(const v8h*)(vp), *(const v8h*)(vp + 16), 0,1,2,3,4,5,6,7,8,9,10,11,12,13,14,15);
        acc[ct] = __builtin_amdgcn_wmma_f32_16x16x32_f16(false, pA, false, vB, (short)0, acc[ct], false, false);
    }

    // merge l across lane halves (same query in lane L and L^16)
    lr += __shfl_xor(lr, 16, 32);

    // per-row gamma/l; acc reg r corresponds to query (r + 8*half)
    const float g = gamma_p[0];
    float linv[8];
    #pragma unroll
    for (int r = 0; r < 8; ++r) linv[r] = g / __shfl(lr, r + 8 * half, 32);

    // writeout: LDS transpose per channel tile -> coalesced rows, fuse +x
    float* outb = out + (size_t)b * C_ * N_;
    const float* xb = x + (size_t)b * C_ * N_;
    for (int ct = 0; ct < 16; ++ct) {
        #pragma unroll
        for (int r = 0; r < 8; ++r)
            stage[l16 * 16 + 8 * half + r] = acc[ct][r] * linv[r];
        __syncthreads();
        int chan = ct * 16 + (lane >> 1);
        int qoff = (lane & 1) * 8;
        const float* s = &stage[(lane >> 1) * 16 + qoff];
        size_t gbase = (size_t)chan * N_ + q0 + qoff;
        #pragma unroll
        for (int i = 0; i < 8; ++i)
            outb[gbase + i] = s[i] + xb[gbase + i];
        __syncthreads();
    }
}

// ---------------------------------------------------------------------------
extern "C" void kernel_launch(void* const* d_in, const int* in_sizes, int n_in,
                              void* d_out, int out_size, void* d_ws, size_t ws_size,
                              hipStream_t stream) {
    (void)in_sizes; (void)n_in; (void)out_size; (void)ws_size;
    const float* x     = (const float*)d_in[0];
    const float* Wq    = (const float*)d_in[1];
    const float* bq    = (const float*)d_in[2];
    const float* Wk    = (const float*)d_in[3];
    const float* bk    = (const float*)d_in[4];
    const float* Wv    = (const float*)d_in[5];
    const float* bv    = (const float*)d_in[6];
    const float* gamma = (const float*)d_in[7];

    _Float16* qh = (_Float16*)d_ws;                 // (B,N,Cq)  f16  1 MB
    _Float16* kh = qh + (size_t)B_ * N_ * CQ_;      // (B,N,Cq)  f16  1 MB
    _Float16* vh = kh + (size_t)B_ * N_ * CQ_;      // (B,C,N)   f16  8 MB

    proj_kernel<<<dim3(N_ / 64, B_), 256, 0, stream>>>(x, Wq, bq, Wk, bk, Wv, bv, qh, kh, vh);
    attn_kernel<<<dim3(N_ / 16, B_), 32, 0, stream>>>(qh, kh, vh, x, gamma, (float*)d_out);
}
